// GPT_52544629899621
// MI455X (gfx1250) — compile-verified, matrix operand order checked by reading, not run on hardware
//
#include <hip/hip_runtime.h>
#include <hip/hip_bf16.h>

// ---------------------------------------------------------------------------
// Decision-Transformer forward for MI455X (gfx1250, wave32, WMMA).
// GEMMs: register-blocked 32x32-per-wave v_wmma_f32_16x16x32_bf16.
// Attention: flash-style, K/V tiles staged via global_load_async_to_lds_b128.
// LN / softmax / GELU / loss stay f32 VALU.
// ---------------------------------------------------------------------------

typedef __attribute__((ext_vector_type(16))) __bf16 v16bf;
typedef __attribute__((ext_vector_type(8)))  float  v8f;

#define BB   16
#define TT   170
#define DD   512
#define SP   512            // padded sequence (real S = 510)
#define SREAL 510
#define HH   8
#define HD   64
#define DFF  2048
#define VV   4672
#define BT   (BB*TT)        // 2720
#define KVP  72             // LDS row pitch for 64-wide K/V tiles (bank spread)

__device__ inline v8f zero8() {
  v8f z;
#pragma unroll
  for (int i = 0; i < 8; ++i) z[i] = 0.0f;
  return z;
}

// K position of the low bf16 of VGPR j for the 16-bit A (and symmetric B)
// fragment layout: lanes 0-15 hold K {0..7,16..23}, lanes 16-31 {8..15,24..31}.
__device__ inline int kpat(int j, int half) {
  return ((j < 4) ? (2 * j) : (16 + 2 * (j - 4))) + 8 * half;
}

// Raw LDS byte offset of a __shared__ object (addrspace(3) ptrtoint).
#define LDS_OFF(p) ((unsigned)(uintptr_t)(__attribute__((address_space(3))) const void*)(p))

// Async global -> LDS copy, 16 bytes per lane (ASYNCcnt-tracked).
__device__ inline void async_g2l_b128(unsigned lds_byte_off, const __bf16* gaddr) {
  asm volatile("global_load_async_to_lds_b128 %0, %1, off"
               :: "v"(lds_byte_off), "v"(gaddr) : "memory");
}
__device__ inline void wait_asynccnt0() {
  asm volatile("s_wait_asynccnt 0x0" ::: "memory");
}

// ---------------------------------------------------------------------------
// f32 -> bf16 conversion
// ---------------------------------------------------------------------------
__global__ __launch_bounds__(256) void f32_to_bf16(const float* __restrict__ s,
                                                   __bf16* __restrict__ d, int n) {
  int i = blockIdx.x * 256 + threadIdx.x;
  if (i < n) d[i] = (__bf16)s[i];
}

// ---------------------------------------------------------------------------
// Register-blocked WMMA GEMM: C[M,N] = act(A[M,K] * W[N,K]^T + bias)
//   Each wave computes a 32x32 output region (2x2 16x16 tiles, 4 WMMA/k-step).
//   Block = 64 threads = 2 waves side by side in N -> block tile 32x64.
//   Grid = (N/64, M/32). Requires M%32==0, N%64==0, K%32==0.
//   outmode: 0 = store f32, 1 = f32 residual add, 2 = store bf16
//   act:     0 = none, 1 = exact GELU
// ---------------------------------------------------------------------------
__global__ __launch_bounds__(64) void gemm_bf16_wmma(
    const __bf16* __restrict__ A, int lda,
    const __bf16* __restrict__ W, int ldw,
    const float* __restrict__ bias,
    float* __restrict__ Cf, __bf16* __restrict__ Cb, int ldc,
    int K, int act, int outmode) {
  const int lane = threadIdx.x & 31;
  const int wave = threadIdx.x >> 5;
  const int m    = lane & 15;
  const int half = lane >> 4;
  const int nt0  = (blockIdx.x * 2 + wave) * 2;  // two 16-col tiles: nt0, nt0+1
  const int mt0  = blockIdx.y * 2;               // two 16-row tiles: mt0, mt0+1

  v8f c00 = zero8(), c01 = zero8(), c10 = zero8(), c11 = zero8();
  const __bf16* ar0 = A + (size_t)(mt0 * 16 + m) * lda;
  const __bf16* ar1 = ar0 + (size_t)16 * lda;
  const __bf16* wr0 = W + (size_t)(nt0 * 16 + m) * ldw;
  const __bf16* wr1 = wr0 + (size_t)16 * ldw;

  for (int k0 = 0; k0 < K; k0 += 32) {
    v16bf a0, a1, w0, w1;
#pragma unroll
    for (int j = 0; j < 8; ++j) {
      const int kb = k0 + kpat(j, half);
      a0[2 * j] = ar0[kb];  a0[2 * j + 1] = ar0[kb + 1];
      a1[2 * j] = ar1[kb];  a1[2 * j + 1] = ar1[kb + 1];
      w0[2 * j] = wr0[kb];  w0[2 * j + 1] = wr0[kb + 1];
      w1[2 * j] = wr1[kb];  w1[2 * j + 1] = wr1[kb + 1];
    }
    __builtin_prefetch(ar0 + k0 + 64, 0, 3);   // WGP-scope global_prefetch_b8
    __builtin_prefetch(wr0 + k0 + 64, 0, 3);
    c00 = __builtin_amdgcn_wmma_f32_16x16x32_bf16(false, a0, false, w0,
                                                  (short)0, c00, false, false);
    c01 = __builtin_amdgcn_wmma_f32_16x16x32_bf16(false, a0, false, w1,
                                                  (short)0, c01, false, false);
    c10 = __builtin_amdgcn_wmma_f32_16x16x32_bf16(false, a1, false, w0,
                                                  (short)0, c10, false, false);
    c11 = __builtin_amdgcn_wmma_f32_16x16x32_bf16(false, a1, false, w1,
                                                  (short)0, c11, false, false);
  }

#pragma unroll
  for (int mi = 0; mi < 2; ++mi) {
#pragma unroll
    for (int ni = 0; ni < 2; ++ni) {
      const v8f& c = (mi == 0) ? (ni == 0 ? c00 : c01) : (ni == 0 ? c10 : c11);
      const int col = (nt0 + ni) * 16 + m;
      const float bv = bias ? bias[col] : 0.0f;
#pragma unroll
      for (int r = 0; r < 8; ++r) {
        const int row = (mt0 + mi) * 16 + r + 8 * half;
        float v = c[r] + bv;
        if (act == 1) v = 0.5f * v * (1.0f + erff(v * 0.70710678118f));
        const size_t o = (size_t)row * ldc + col;
        if (outmode == 0)      Cf[o] = v;
        else if (outmode == 1) Cf[o] += v;
        else                   Cb[o] = (__bf16)v;
      }
    }
  }
}

// ---------------------------------------------------------------------------
// Flash attention, one wave per (q-tile of 16, head, batch). S padded to 512,
// key columns >= 510 masked. K/V tiles async-staged to LDS; scores and P*V
// on WMMA; online softmax f32.
// ---------------------------------------------------------------------------
__global__ __launch_bounds__(32) void attn_flash(
    const __bf16* __restrict__ Q, const __bf16* __restrict__ K,
    const __bf16* __restrict__ V, __bf16* __restrict__ Y) {
  const int qt = blockIdx.x;   // 0..31
  const int hh = blockIdx.y;   // 0..7
  const int b  = blockIdx.z;   // 0..15
  const int lane = threadIdx.x;
  const int m    = lane & 15;
  const int half = lane >> 4;

  __shared__ __bf16 kt_s[32 * KVP];   // K tile: 32 keys x 64 dims (pitch 72)
  __shared__ __bf16 vt_s[32 * KVP];   // V tile
  __shared__ __bf16 pls[16][34];      // P tile bounce (C-layout -> A-layout)
  const unsigned kt_base = LDS_OFF(kt_s);
  const unsigned vt_base = LDS_OFF(vt_s);

  const size_t base = ((size_t)b * SP) * DD + (size_t)hh * HD;
  const __bf16* Qb = Q + base;
  const __bf16* Kb = K + base;
  const __bf16* Vb = V + base;

  // Q fragments for K-dim 0..31 and 32..63 (loaded once, direct global)
  v16bf aq0, aq1;
  {
    const __bf16* qp = Qb + (size_t)(qt * 16 + m) * DD;
#pragma unroll
    for (int j = 0; j < 8; ++j) {
      const int kb = kpat(j, half);
      aq0[2 * j]     = qp[kb];
      aq0[2 * j + 1] = qp[kb + 1];
      aq1[2 * j]     = qp[32 + kb];
      aq1[2 * j + 1] = qp[32 + kb + 1];
    }
  }

  v8f o0 = zero8(), o1 = zero8(), o2 = zero8(), o3 = zero8();
  float mi[8], li[8];
#pragma unroll
  for (int r = 0; r < 8; ++r) { mi[r] = -1e30f; li[r] = 0.0f; }

  // cooperative copy geometry: 8 lanes x 16B cover one 128B (64xbf16) row,
  // 32 lanes move 4 rows per issue, 8 issues cover the 32-row tile.
  const int crow = lane >> 3;          // 0..3
  const int coff = (lane & 7) * 8;     // bf16 element offset within row

  const int ktend = (qt * 16 + 15) >> 5;   // causal: last 32-key step needed
  for (int kt = 0; kt <= ktend; ++kt) {
    const int kk = kt * 32;

    // ---- async stage K/V tiles into LDS (ASYNCcnt-tracked) ----
#pragma unroll
    for (int i = 0; i < 8; ++i) {
      const int row = i * 4 + crow;
      const unsigned lof = (unsigned)(row * KVP + coff) * 2u;
      const __bf16* gk = Kb + (size_t)(kk + row) * DD + coff;
      const __bf16* gv = Vb + (size_t)(kk + row) * DD + coff;
      async_g2l_b128(kt_base + lof, gk);
      async_g2l_b128(vt_base + lof, gv);
    }
    wait_asynccnt0();
    __syncthreads();

    // ---- scores: S = Q K^T (WMMA, B-fragments from LDS) ----
    v8f s0 = zero8(), s1 = zero8();
    {
      const __bf16* kp = kt_s + (size_t)m * KVP;          // keys kk..kk+15
      v16bf b0, b1;
#pragma unroll
      for (int j = 0; j < 8; ++j) {
        const int kb = kpat(j, half);
        b0[2 * j] = kp[kb];      b0[2 * j + 1] = kp[kb + 1];
        b1[2 * j] = kp[32 + kb]; b1[2 * j + 1] = kp[32 + kb + 1];
      }
      s0 = __builtin_amdgcn_wmma_f32_16x16x32_bf16(false, aq0, false, b0,
                                                   (short)0, s0, false, false);
      s0 = __builtin_amdgcn_wmma_f32_16x16x32_bf16(false, aq1, false, b1,
                                                   (short)0, s0, false, false);
    }
    {
      const __bf16* kp = kt_s + (size_t)(16 + m) * KVP;   // keys kk+16..kk+31
      v16bf b0, b1;
#pragma unroll
      for (int j = 0; j < 8; ++j) {
        const int kb = kpat(j, half);
        b0[2 * j] = kp[kb];      b0[2 * j + 1] = kp[kb + 1];
        b1[2 * j] = kp[32 + kb]; b1[2 * j + 1] = kp[32 + kb + 1];
      }
      s1 = __builtin_amdgcn_wmma_f32_16x16x32_bf16(false, aq0, false, b0,
                                                   (short)0, s1, false, false);
      s1 = __builtin_amdgcn_wmma_f32_16x16x32_bf16(false, aq1, false, b1,
                                                   (short)0, s1, false, false);
    }

    // ---- scale + causal/pad mask + online softmax (f32) ----
#pragma unroll
    for (int r = 0; r < 8; ++r) {
      const int qrow = qt * 16 + r + 8 * half;
      const int c0 = kk + m, c1 = kk + 16 + m;
      float v0 = s0[r] * 0.125f;   // 1/sqrt(64)
      float v1 = s1[r] * 0.125f;
      if (c0 > qrow || c0 >= SREAL) v0 = -1e30f;
      if (c1 > qrow || c1 >= SREAL) v1 = -1e30f;
      float rmax = fmaxf(v0, v1);
      rmax = fmaxf(rmax, __shfl_xor(rmax, 1, 32));
      rmax = fmaxf(rmax, __shfl_xor(rmax, 2, 32));
      rmax = fmaxf(rmax, __shfl_xor(rmax, 4, 32));
      rmax = fmaxf(rmax, __shfl_xor(rmax, 8, 32));
      const float mnew = fmaxf(mi[r], rmax);
      const float corr = __expf(mi[r] - mnew);
      const float p0 = __expf(v0 - mnew);
      const float p1 = __expf(v1 - mnew);
      float psum = p0 + p1;
      psum += __shfl_xor(psum, 1, 32);
      psum += __shfl_xor(psum, 2, 32);
      psum += __shfl_xor(psum, 4, 32);
      psum += __shfl_xor(psum, 8, 32);
      li[r] = li[r] * corr + psum;
      mi[r] = mnew;
      o0[r] *= corr; o1[r] *= corr; o2[r] *= corr; o3[r] *= corr;
      pls[r + 8 * half][m]      = (__bf16)p0;
      pls[r + 8 * half][16 + m] = (__bf16)p1;
    }
    __syncthreads();

    // reload P as A-fragment (16x32) from LDS
    v16bf ap;
#pragma unroll
    for (int j = 0; j < 8; ++j) {
      const int kb = kpat(j, half);
      ap[2 * j]     = pls[m][kb];
      ap[2 * j + 1] = pls[m][kb + 1];
    }
    // ---- o += P @ V (4 column tiles of 16 over head dim, V from LDS) ----
#pragma unroll
    for (int t = 0; t < 4; ++t) {
      v16bf bv;
#pragma unroll
      for (int j = 0; j < 8; ++j) {
        const int kb = kpat(j, half);
        bv[2 * j]     = vt_s[(size_t)kb       * KVP + t * 16 + m];
        bv[2 * j + 1] = vt_s[(size_t)(kb + 1) * KVP + t * 16 + m];
      }
      v8f* op = (t == 0) ? &o0 : (t == 1) ? &o1 : (t == 2) ? &o2 : &o3;
      *op = __builtin_amdgcn_wmma_f32_16x16x32_bf16(false, ap, false, bv,
                                                    (short)0, *op, false, false);
    }
    __syncthreads();
  }

  // normalize and store bf16
#pragma unroll
  for (int r = 0; r < 8; ++r) {
    const int row = qt * 16 + r + 8 * half;
    const float inv = 1.0f / li[r];
    __bf16* yp = Y + ((size_t)b * SP + row) * DD + hh * HD;
    yp[0  + m] = (__bf16)(o0[r] * inv);
    yp[16 + m] = (__bf16)(o1[r] * inv);
    yp[32 + m] = (__bf16)(o2[r] * inv);
    yp[48 + m] = (__bf16)(o3[r] * inv);
  }
}

// ---------------------------------------------------------------------------
// LayerNorm (f32 in -> bf16 out), one block per row of 512
// ---------------------------------------------------------------------------
__global__ __launch_bounds__(256) void ln_to_bf16(
    const float* __restrict__ X, const float* __restrict__ g,
    const float* __restrict__ bb, __bf16* __restrict__ O) {
  __shared__ float s1[256], s2[256];
  const size_t row = blockIdx.x;
  const float* xr = X + row * DD;
  const int tid = threadIdx.x;
  const float a  = xr[tid];
  const float b2 = xr[tid + 256];
  s1[tid] = a + b2;
  s2[tid] = a * a + b2 * b2;
  __syncthreads();
  for (int s = 128; s > 0; s >>= 1) {
    if (tid < s) { s1[tid] += s1[tid + s]; s2[tid] += s2[tid + s]; }
    __syncthreads();
  }
  const float mean = s1[0] * (1.0f / DD);
  const float var  = s2[0] * (1.0f / DD) - mean * mean;
  const float rs   = rsqrtf(var + 1e-5f);
  O[row * DD + tid]       = (__bf16)((a  - mean) * rs * g[tid]       + bb[tid]);
  O[row * DD + tid + 256] = (__bf16)((b2 - mean) * rs * g[tid + 256] + bb[tid + 256]);
}

// ---------------------------------------------------------------------------
// Gather action-token rows (s = 3t+2), final LayerNorm -> bf16 compact rows
// ---------------------------------------------------------------------------
__global__ __launch_bounds__(256) void gather_lnf(
    const float* __restrict__ X, const float* __restrict__ g,
    const float* __restrict__ bb, __bf16* __restrict__ O) {
  __shared__ float s1[256], s2[256];
  const int row = blockIdx.x;                 // 0..BT-1
  const int b = row / TT, t = row - b * TT;
  const float* xr = X + ((size_t)b * SP + 3 * t + 2) * DD;
  const int tid = threadIdx.x;
  const float a  = xr[tid];
  const float b2 = xr[tid + 256];
  s1[tid] = a + b2;
  s2[tid] = a * a + b2 * b2;
  __syncthreads();
  for (int s = 128; s > 0; s >>= 1) {
    if (tid < s) { s1[tid] += s1[tid + s]; s2[tid] += s2[tid + s]; }
    __syncthreads();
  }
  const float mean = s1[0] * (1.0f / DD);
  const float var  = s2[0] * (1.0f / DD) - mean * mean;
  const float rs   = rsqrtf(var + 1e-5f);
  O[(size_t)row * DD + tid]       = (__bf16)((a  - mean) * rs * g[tid]       + bb[tid]);
  O[(size_t)row * DD + tid + 256] = (__bf16)((b2 - mean) * rs * g[tid + 256] + bb[tid + 256]);
}

// ---------------------------------------------------------------------------
// Token assembly: x = stack(tanh(goal_lin), state_lin, tanh(act_emb)) + pos
// Rows s >= 510 zero-padded.
// ---------------------------------------------------------------------------
__global__ __launch_bounds__(256) void assemble_x(
    const float* __restrict__ st_f, const int* __restrict__ actions,
    const float* __restrict__ goals, const int* __restrict__ timesteps,
    const float* __restrict__ act_emb, const float* __restrict__ goal_W,
    const float* __restrict__ goal_b, const float* __restrict__ pos_emb,
    const float* __restrict__ gpos_emb, float* __restrict__ X) {
  const size_t idx = (size_t)blockIdx.x * 256 + threadIdx.x;  // < B*SP*D
  const int d = (int)(idx & 511);
  const int s = (int)((idx >> 9) & 511);
  const int b = (int)(idx >> 18);
  if (s >= SREAL) { X[idx] = 0.0f; return; }
  const int t = s / 3, r = s - 3 * t;
  const int bt = b * TT + t;
  float tok;
  if (r == 0)      tok = tanhf(goals[bt] * goal_W[d] + goal_b[d]);
  else if (r == 1) tok = st_f[(size_t)bt * DD + d];
  else             tok = tanhf(act_emb[(size_t)actions[bt] * DD + d]);
  X[idx] = tok + gpos_emb[(size_t)timesteps[bt] * DD + d]
               + pos_emb[(size_t)s * DD + d];
}

// ---------------------------------------------------------------------------
// Per-row log-softmax NLL, mean over B*T accumulated atomically.
// ---------------------------------------------------------------------------
__global__ __launch_bounds__(256) void loss_kernel(
    const float* __restrict__ logits, const int* __restrict__ targets,
    float* __restrict__ loss) {
  __shared__ float red[256];
  const int row = blockIdx.x;
  const float* lr = logits + (size_t)row * VV;
  const int tid = threadIdx.x;
  float mx = -3.0e38f;
  for (int i = tid; i < VV; i += 256) mx = fmaxf(mx, lr[i]);
  red[tid] = mx; __syncthreads();
  for (int s = 128; s > 0; s >>= 1) {
    if (tid < s) red[tid] = fmaxf(red[tid], red[tid + s]);
    __syncthreads();
  }
  mx = red[0]; __syncthreads();
  float sum = 0.0f;
  for (int i = tid; i < VV; i += 256) sum += __expf(lr[i] - mx);
  red[tid] = sum; __syncthreads();
  for (int s = 128; s > 0; s >>= 1) {
    if (tid < s) red[tid] += red[tid + s];
    __syncthreads();
  }
  if (tid == 0) {
    const float lse = mx + __logf(red[0]);
    atomicAdd(loss, (lse - lr[targets[row]]) * (1.0f / (float)BT));
  }
}

// ---------------------------------------------------------------------------
// Host orchestration
// ---------------------------------------------------------------------------
extern "C" void kernel_launch(void* const* d_in, const int* in_sizes, int n_in,
                              void* d_out, int out_size, void* d_ws, size_t ws_size,
                              hipStream_t stream) {
  (void)in_sizes; (void)n_in; (void)out_size; (void)ws_size;

  const float* states    = (const float*)d_in[0];
  const int*   actions   = (const int*)d_in[1];
  const float* goals     = (const float*)d_in[2];
  const int*   timesteps = (const int*)d_in[3];
  const int*   targets   = (const int*)d_in[4];
  const float* state_W   = (const float*)d_in[5];
  const float* state_b   = (const float*)d_in[6];
  const float* act_emb   = (const float*)d_in[7];
  const float* goal_W    = (const float*)d_in[8];
  const float* goal_b    = (const float*)d_in[9];
  const float* pos_emb   = (const float*)d_in[10];
  const float* gpos_emb  = (const float*)d_in[11];
  const float* ln1_g = (const float*)d_in[12];
  const float* ln1_b = (const float*)d_in[13];
  const float* Wq = (const float*)d_in[14]; const float* bq = (const float*)d_in[15];
  const float* Wk = (const float*)d_in[16]; const float* bk = (const float*)d_in[17];
  const float* Wv = (const float*)d_in[18]; const float* bv = (const float*)d_in[19];
  const float* Wp = (const float*)d_in[20]; const float* bp = (const float*)d_in[21];
  const float* ln2_g = (const float*)d_in[22]; const float* ln2_b = (const float*)d_in[23];
  const float* W1 = (const float*)d_in[24]; const float* b1 = (const float*)d_in[25];
  const float* W2 = (const float*)d_in[26]; const float* b2 = (const float*)d_in[27];
  const float* lnf_g = (const float*)d_in[28]; const float* lnf_b = (const float*)d_in[29];
  const float* head_W = (const float*)d_in[30];

  float* logits = (float*)d_out;                       // BT x V
  float* loss   = logits + (size_t)BT * VV;            // scalar

  // --- bump allocator over d_ws ---
  size_t off = 0;
  auto alloc = [&](size_t bytes) -> void* {
    void* p = (char*)d_ws + off;
    off += (bytes + 255) & ~(size_t)255;
    return p;
  };
  __bf16* states_bf = (__bf16*)alloc((size_t)BT * 768 * 2);
  __bf16* sW_bf     = (__bf16*)alloc((size_t)DD * 768 * 2);
  __bf16* wq_bf     = (__bf16*)alloc((size_t)DD * DD * 2);
  __bf16* wk_bf     = (__bf16*)alloc((size_t)DD * DD * 2);
  __bf16* wv_bf     = (__bf16*)alloc((size_t)DD * DD * 2);
  __bf16* wp_bf     = (__bf16*)alloc((size_t)DD * DD * 2);
  __bf16* w1_bf     = (__bf16*)alloc((size_t)DFF * DD * 2);
  __bf16* w2_bf     = (__bf16*)alloc((size_t)DD * DFF * 2);
  __bf16* hW_bf     = (__bf16*)alloc((size_t)VV * DD * 2);
  __bf16* hbf       = (__bf16*)alloc((size_t)BB * SP * DD * 2);  // LN out / attn Y reuse
  __bf16* qbf       = (__bf16*)alloc((size_t)BB * SP * DD * 2);
  __bf16* kbf       = (__bf16*)alloc((size_t)BB * SP * DD * 2);
  __bf16* vbf       = (__bf16*)alloc((size_t)BB * SP * DD * 2);
  __bf16* gbf       = (__bf16*)alloc((size_t)BB * SP * DFF * 2);
  __bf16* xg_bf     = (__bf16*)alloc((size_t)BT * DD * 2);
  float*  x_f       = (float*)alloc((size_t)BB * SP * DD * 4);
  float*  st_f      = (float*)alloc((size_t)BT * DD * 4);

  auto conv = [&](const float* s, __bf16* dp, int n) {
    f32_to_bf16<<<(n + 255) / 256, 256, 0, stream>>>(s, dp, n);
  };
  auto gemm = [&](const __bf16* A, int lda, const __bf16* Wt, int ldw,
                  const float* bias, float* Cf, __bf16* Cb, int ldc,
                  int M, int N, int K, int act, int outmode) {
    dim3 g(N / 64, M / 32);
    gemm_bf16_wmma<<<g, 64, 0, stream>>>(A, lda, Wt, ldw, bias, Cf, Cb, ldc,
                                         K, act, outmode);
  };

  hipMemsetAsync(loss, 0, sizeof(float), stream);

  // --- embeddings ---
  conv(states, states_bf, BT * 768);
  conv(state_W, sW_bf, DD * 768);
  conv(head_W, hW_bf, VV * DD);
  gemm(states_bf, 768, sW_bf, 768, state_b, st_f, nullptr, DD,
       BT, DD, 768, 0, 0);
  assemble_x<<<(BB * SP * DD) / 256, 256, 0, stream>>>(
      st_f, actions, goals, timesteps, act_emb, goal_W, goal_b,
      pos_emb, gpos_emb, x_f);

  // --- transformer blocks ---
  const int MR = BB * SP;  // 8192 padded rows
  for (int l = 0; l < 8; ++l) {
    conv(Wq + (size_t)l * DD * DD, wq_bf, DD * DD);
    conv(Wk + (size_t)l * DD * DD, wk_bf, DD * DD);
    conv(Wv + (size_t)l * DD * DD, wv_bf, DD * DD);
    conv(Wp + (size_t)l * DD * DD, wp_bf, DD * DD);
    conv(W1 + (size_t)l * DFF * DD, w1_bf, DFF * DD);
    conv(W2 + (size_t)l * DD * DFF, w2_bf, DD * DFF);

    ln_to_bf16<<<MR, 256, 0, stream>>>(x_f, ln1_g + l * DD, ln1_b + l * DD, hbf);
    gemm(hbf, DD, wq_bf, DD, bq + l * DD, nullptr, qbf, DD, MR, DD, DD, 0, 2);
    gemm(hbf, DD, wk_bf, DD, bk + l * DD, nullptr, kbf, DD, MR, DD, DD, 0, 2);
    gemm(hbf, DD, wv_bf, DD, bv + l * DD, nullptr, vbf, DD, MR, DD, DD, 0, 2);

    attn_flash<<<dim3(SP / 16, HH, BB), 32, 0, stream>>>(qbf, kbf, vbf, hbf);

    gemm(hbf, DD, wp_bf, DD, bp + l * DD, x_f, nullptr, DD, MR, DD, DD, 0, 1);

    ln_to_bf16<<<MR, 256, 0, stream>>>(x_f, ln2_g + l * DD, ln2_b + l * DD, hbf);
    gemm(hbf, DD, w1_bf, DD, b1 + l * DFF, nullptr, gbf, DFF, MR, DFF, DD, 1, 2);
    gemm(gbf, DFF, w2_bf, DFF, b2 + l * DD, x_f, nullptr, DD, MR, DD, DFF, 0, 1);
  }

  // --- head + loss ---
  gather_lnf<<<BT, 256, 0, stream>>>(x_f, lnf_g, lnf_b, xg_bf);
  gemm(xg_bf, DD, hW_bf, DD, nullptr, logits, nullptr, VV, BT, VV, DD, 0, 0);
  loss_kernel<<<BT, 256, 0, stream>>>(logits, targets, loss);
}
